// Model_Recursive_LSTM_v2_2594160247426
// MI455X (gfx1250) — matrix operand order repaired and checked by reading, not compiled, not run on hardware
//
#include <hip/hip_runtime.h>
#include <hip/hip_bf16.h>
#include <stdint.h>

// ---------------- CDNA5 WMMA types ----------------
typedef __attribute__((ext_vector_type(16))) __bf16 v16bf;
typedef __attribute__((ext_vector_type(8)))  float  v8f;

#define TM 128
#define TN 64

__device__ __forceinline__ float sigf(float x) { return 1.0f / (1.0f + expf(-x)); }

// ---------------- Generic WMMA GEMM ----------------
// C[M x N] = act( A[M x K](bf16) @ W[N x K]^T(bf16) + bias[N] )
// ACT==1: ELU, store bf16.  ACT==0: none, store f32.
// M % 128 == 0, N % 64 == 0, K % 32 == 0 (all padded).
// Block: 256 threads = 8 waves in 4(M) x 2(N); each wave owns a 32x32 tile
// (4 v_wmma accumulators). LDS double-buffered: one barrier per K-step.
template <int ACT>
__global__ __launch_bounds__(256) void k_gemm(const __bf16* __restrict__ A,
                                              const __bf16* __restrict__ W,
                                              const float* __restrict__ bias,
                                              void* __restrict__ Cout,
                                              int M, int N, int K) {
  __shared__ __align__(16) uint32_t sA[2][TM * 16];  // 128 rows x 32 bf16
  __shared__ __align__(16) uint32_t sB[2][TN * 16];  // 64 W-rows x 32 bf16

  const int tid  = threadIdx.x;
  const int bm   = blockIdx.x;
  const int bn   = blockIdx.y;
  const int wave = tid >> 5;
  const int lane = tid & 31;
  const int wm   = wave >> 1;  // 0..3 -> 32-row M sub-tile
  const int wn   = wave & 1;   // 0..1 -> 32-col N sub-tile

  const uint32_t* Au = (const uint32_t*)A;
  const uint32_t* Wu = (const uint32_t*)W;
  const int KD = K >> 1;  // dwords per row

  // Cooperative loaders: A tile = 512 uint4 (2/thread), B tile = 256 uint4.
  const int ua0r = tid >> 2, ua0q = (tid & 3) << 2;
  const int ua1r = (tid + 256) >> 2, ua1q = ua0q;  // (u&3) identical for +256
  uint4 r0, r1, r2;
  auto gload = [&](int st) {
    const int kd = st << 4;  // 16 dwords per K-step
    r0 = *(const uint4*)(Au + (size_t)(bm * TM + ua0r) * KD + kd + ua0q);
    r1 = *(const uint4*)(Au + (size_t)(bm * TM + ua1r) * KD + kd + ua1q);
    r2 = *(const uint4*)(Wu + (size_t)(bn * TN + ua0r) * KD + kd + ua0q);
  };
  auto lstore = [&](int buf) {
    *(uint4*)&sA[buf][ua0r * 16 + ua0q] = r0;
    *(uint4*)&sA[buf][ua1r * 16 + ua1q] = r1;
    *(uint4*)&sB[buf][ua0r * 16 + ua0q] = r2;
  };

  v8f acc[2][2] = {};

  gload(0);
  lstore(0);
  __syncthreads();

  const int nsteps = K >> 5;
  const int g  = lane >> 4;    // K-half select
  const int ml = lane & 15;    // M (or N) row within 16

  for (int st = 0; st < nsteps; ++st) {
    const int cur = st & 1;
    const bool more = (st + 1) < nsteps;
    if (more) gload(st + 1);

    // A fragments (16x32 each): VGPR v -> dword hi*8 + g*4 + vv
    union { uint32_t u[8]; v16bf v; } af[2], bfr[2];
#pragma unroll
    for (int mt = 0; mt < 2; ++mt) {
      const int mrow = wm * 32 + mt * 16 + ml;
#pragma unroll
      for (int v = 0; v < 8; ++v) {
        const int vv = v & 3, hi = v >> 2;
        af[mt].u[v] = sA[cur][mrow * 16 + hi * 8 + g * 4 + vv];
      }
    }
    // B fragments (32x16 each): VGPR v -> dword (v>>2)*8 + g*4 + (v&3)
#pragma unroll
    for (int nt = 0; nt < 2; ++nt) {
      const int nrow = wn * 32 + nt * 16 + ml;
#pragma unroll
      for (int v = 0; v < 8; ++v) bfr[nt].u[v] = sB[cur][nrow * 16 + (v >> 2) * 8 + g * 4 + (v & 3)];
    }
#pragma unroll
    for (int mt = 0; mt < 2; ++mt)
#pragma unroll
      for (int nt = 0; nt < 2; ++nt)
        acc[mt][nt] = __builtin_amdgcn_wmma_f32_16x16x32_bf16(
            false, af[mt].v, false, bfr[nt].v, (short)0, acc[mt][nt], false, false);

    if (more) lstore(cur ^ 1);
    __syncthreads();
  }

  // Epilogue. C/D layout: VGPR r -> M = r (lanes 0-15) or 8+r (lanes 16-31),
  // N = lane & 15.
#pragma unroll
  for (int mt = 0; mt < 2; ++mt) {
    const int rb = bm * TM + wm * 32 + mt * 16 + (g << 3);
#pragma unroll
    for (int nt = 0; nt < 2; ++nt) {
      const int col = bn * TN + wn * 32 + nt * 16 + ml;
      const float bv = bias[col];
#pragma unroll
      for (int r = 0; r < 8; ++r) {
        const int row = rb + r;
        float x = acc[mt][nt][r] + bv;
        if (ACT == 1) {
          x = x > 0.0f ? x : (expf(x) - 1.0f);
          ((__bf16*)Cout)[(size_t)row * N + col] = (__bf16)x;
        } else {
          ((float*)Cout)[(size_t)row * N + col] = x;
        }
      }
    }
  }
}

// ---------------- Packing / conversion kernels ----------------
// Pack f32 weight (N x K) into zeroed bf16 (rows x Kp); src col k goes to
// dst col (k<split ? k+off0 : k-split+off1).
__global__ void k_pack_w(const float* __restrict__ src, __bf16* __restrict__ dst,
                         int N, int K, int Kp, int split, int off0, int off1) {
  int i = blockIdx.x * blockDim.x + threadIdx.x;
  if (i >= N * K) return;
  int n = i / K, k = i - n * K;
  int dc = (k < split) ? (k + off0) : (k - split + off1);
  dst[(size_t)n * Kp + dc] = (__bf16)src[i];
}

__global__ void k_pack_b(const float* __restrict__ src, float* __restrict__ dst, int N) {
  int i = blockIdx.x * blockDim.x + threadIdx.x;
  if (i < N) dst[i] = src[i];
}

__global__ void k_add_b(const float* __restrict__ a, const float* __restrict__ b,
                        float* __restrict__ dst, int N) {
  int i = blockIdx.x * blockDim.x + threadIdx.x;
  if (i < N) dst[i] = a[i] + b[i];
}

__global__ void k_cvt_bf16(const float* __restrict__ src, __bf16* __restrict__ dst, int n) {
  int i = blockIdx.x * blockDim.x + threadIdx.x;
  if (i < n) dst[i] = (__bf16)src[i];
}

// ---------------- LSTM step helpers ----------------
// Build [x_t | h] (Bx x 384 bf16). mode 0: leaf (x from embed, row=b*64+8j+t),
// mode 1: interior (x from leaf states, row=(2i+t)*256+b), mode 2: root
// (x from mid states, row=t*256+n). blockDim=192, grid=Bx.
__global__ void k_build_xh(const __bf16* __restrict__ X, const __bf16* __restrict__ H,
                           __bf16* __restrict__ out, int mode, int t) {
  int n = blockIdx.x, c = threadIdx.x;
  int b = n & 255, j = n >> 8;
  int row;
  if (mode == 0)      row = b * 64 + 8 * j + t;
  else if (mode == 1) row = (2 * j + t) * 256 + b;
  else                row = t * 256 + n;
  out[(size_t)n * 384 + c]       = X[(size_t)row * 192 + c];
  out[(size_t)n * 384 + 192 + c] = H[(size_t)n * 192 + c];
}

// Build concat input [left(192) | right(192)] where each side is a per-row
// bf16 matrix or a broadcast f32 row (180 real values). blockDim=192, grid=Bx.
__global__ void k_build_concat(const __bf16* __restrict__ Lmat, const float* __restrict__ Lrow,
                               const __bf16* __restrict__ Rmat, const float* __restrict__ Rrow,
                               __bf16* __restrict__ out) {
  int n = blockIdx.x, c = threadIdx.x;
  __bf16 lv = Lmat ? Lmat[(size_t)n * 192 + c] : (__bf16)((c < 180) ? Lrow[c] : 0.0f);
  __bf16 rv = Rmat ? Rmat[(size_t)n * 192 + c] : (__bf16)((c < 180) ? Rrow[c] : 0.0f);
  out[(size_t)n * 384 + c]       = lv;
  out[(size_t)n * 384 + 192 + c] = rv;
}

// LSTM cell update from gates (Bx x 768 f32, torch order i,f,g,o each 180).
// blockDim=192, grid=Bx.
__global__ void k_lstm_update(const float* __restrict__ gates, float* __restrict__ c,
                              __bf16* __restrict__ h) {
  int n = blockIdx.x, e = threadIdx.x;
  float hv = 0.0f, cv = 0.0f;
  if (e < 180) {
    size_t gb = (size_t)n * 768;
    float gi = sigf(gates[gb + e]);
    float gf = sigf(gates[gb + 180 + e]);
    float gg = tanhf(gates[gb + 360 + e]);
    float go = sigf(gates[gb + 540 + e]);
    cv = gf * c[(size_t)n * 192 + e] + gi * gg;
    hv = go * tanhf(cv);
  }
  c[(size_t)n * 192 + e] = cv;
  h[(size_t)n * 192 + e] = (__bf16)hv;
}

// Final prediction: out[n] = x[n,:180] . pred_w + pred_b
__global__ void k_pred(const __bf16* __restrict__ x, const float* __restrict__ w,
                       const float* __restrict__ b, float* __restrict__ out) {
  int n = blockIdx.x * blockDim.x + threadIdx.x;
  if (n >= 256) return;
  float s = b[0];
  for (int e = 0; e < 180; ++e) s += (float)x[(size_t)n * 192 + e] * w[e];
  out[n] = s;
}

// ---------------- Host orchestration ----------------
extern "C" void kernel_launch(void* const* d_in, const int* /*in_sizes*/, int /*n_in*/,
                              void* d_out, int /*out_size*/, void* d_ws, size_t /*ws_size*/,
                              hipStream_t stream) {
  const float* ast = (const float*)d_in[0];
  const float* ce_w[4] = {(const float*)d_in[1], (const float*)d_in[3],
                          (const float*)d_in[5], (const float*)d_in[7]};
  const float* ce_b[4] = {(const float*)d_in[2], (const float*)d_in[4],
                          (const float*)d_in[6], (const float*)d_in[8]};
  const float* cc_w[2] = {(const float*)d_in[9], (const float*)d_in[11]};
  const float* cc_b[2] = {(const float*)d_in[10], (const float*)d_in[12]};
  const float* rg_w[2] = {(const float*)d_in[13], (const float*)d_in[15]};
  const float* rg_b[2] = {(const float*)d_in[14], (const float*)d_in[16]};
  const float* pred_w  = (const float*)d_in[17];
  const float* pred_b  = (const float*)d_in[18];
  const float* cl_Wih  = (const float*)d_in[19];
  const float* cl_Whh  = (const float*)d_in[20];
  const float* cl_bih  = (const float*)d_in[21];
  const float* cl_bhh  = (const float*)d_in[22];
  const float* nl_Wih  = (const float*)d_in[23];
  const float* nl_Whh  = (const float*)d_in[24];
  const float* nl_bih  = (const float*)d_in[25];
  const float* nl_bhh  = (const float*)d_in[26];
  const float* no_comps = (const float*)d_in[27];
  const float* no_nodes = (const float*)d_in[28];

  char* ws = (char*)d_ws;
  size_t off = 0;
  auto alloc = [&](size_t b) { size_t o = off; off += (b + 255) & ~(size_t)255; return o; };

  // Padded dims: MLP 1024 -> 640 -> 384 -> 256 -> 192 ; E->192 ; gates->768.
  const size_t ML = (size_t)16384;  // B*L rows
  size_t oEmbed = alloc(ML * 192 * 2);
  size_t oBig0  = alloc(ML * 1024 * 2);  // bf16 ast; later LSTM arena
  size_t oBig1  = alloc(ML * 640 * 2);   // a1, later a3
  size_t oBig2  = alloc(ML * 384 * 2);   // a2
  size_t wBeg = off;
  size_t oWce0 = alloc((size_t)640 * 1024 * 2);
  size_t oWce1 = alloc((size_t)384 * 640 * 2);
  size_t oWce2 = alloc((size_t)256 * 384 * 2);
  size_t oWce3 = alloc((size_t)192 * 256 * 2);
  size_t oWcc0 = alloc((size_t)256 * 384 * 2);
  size_t oWcc1 = alloc((size_t)192 * 256 * 2);
  size_t oWrg0 = alloc((size_t)256 * 192 * 2);
  size_t oWrg1 = alloc((size_t)192 * 256 * 2);
  size_t oWcl  = alloc((size_t)768 * 384 * 2);
  size_t oWnl  = alloc((size_t)768 * 384 * 2);
  size_t oBce0 = alloc(640 * 4);
  size_t oBce1 = alloc(384 * 4);
  size_t oBce2 = alloc(256 * 4);
  size_t oBce3 = alloc(192 * 4);
  size_t oBcc0 = alloc(256 * 4);
  size_t oBcc1 = alloc(192 * 4);
  size_t oBrg0 = alloc(256 * 4);
  size_t oBrg1 = alloc(192 * 4);
  size_t oBcl  = alloc(768 * 4);
  size_t oBnl  = alloc(768 * 4);
  size_t wEnd = off;

  // Small LSTM arena reuses Big0 (free after MLP layer 1).
  size_t s = oBig0;
  auto salloc = [&](size_t b) { size_t o = s; s += (b + 255) & ~(size_t)255; return o; };
  size_t oXh    = salloc((size_t)1536 * 384 * 2);
  size_t oGates = salloc((size_t)1536 * 768 * 4);
  size_t oH     = salloc((size_t)1536 * 192 * 2);
  size_t oC     = salloc((size_t)1536 * 192 * 4);
  size_t oLeaf  = salloc((size_t)1536 * 192 * 2);
  size_t oCct   = salloc((size_t)1536 * 256 * 2);
  size_t oMid   = salloc((size_t)768 * 192 * 2);
  size_t oProg  = salloc((size_t)256 * 192 * 2);
  size_t oRga   = salloc((size_t)256 * 256 * 2);
  size_t oRgb   = salloc((size_t)256 * 192 * 2);

  auto BF = [&](size_t o) { return (__bf16*)(ws + o); };
  auto FP = [&](size_t o) { return (float*)(ws + o); };

  // ---- Pack weights (zero padded regions first) ----
  hipMemsetAsync(ws + wBeg, 0, wEnd - wBeg, stream);
  auto packW = [&](const float* src, size_t dst, int N, int K, int Kp, int split,
                   int o0, int o1) {
    int n = N * K;
    k_pack_w<<<(n + 255) / 256, 256, 0, stream>>>(src, BF(dst), N, K, Kp, split, o0, o1);
  };
  packW(ce_w[0], oWce0, 600, 1024, 1024, 1024, 0, 0);
  packW(ce_w[1], oWce1, 350, 600, 640, 600, 0, 0);
  packW(ce_w[2], oWce2, 200, 350, 384, 350, 0, 0);
  packW(ce_w[3], oWce3, 180, 200, 256, 200, 0, 0);
  packW(cc_w[0], oWcc0, 200, 360, 384, 180, 0, 192);  // [nodes|comps] split
  packW(cc_w[1], oWcc1, 180, 200, 256, 200, 0, 0);
  packW(rg_w[0], oWrg0, 200, 180, 192, 180, 0, 0);
  packW(rg_w[1], oWrg1, 180, 200, 256, 200, 0, 0);
  packW(cl_Wih, oWcl, 720, 180, 384, 180, 0, 0);   // cols 0..179
  packW(cl_Whh, oWcl, 720, 180, 384, 180, 192, 0); // cols 192..371
  packW(nl_Wih, oWnl, 720, 180, 384, 180, 0, 0);
  packW(nl_Whh, oWnl, 720, 180, 384, 180, 192, 0);
  k_pack_b<<<3, 256, 0, stream>>>(ce_b[0], FP(oBce0), 600);
  k_pack_b<<<2, 256, 0, stream>>>(ce_b[1], FP(oBce1), 350);
  k_pack_b<<<1, 256, 0, stream>>>(ce_b[2], FP(oBce2), 200);
  k_pack_b<<<1, 256, 0, stream>>>(ce_b[3], FP(oBce3), 180);
  k_pack_b<<<1, 256, 0, stream>>>(cc_b[0], FP(oBcc0), 200);
  k_pack_b<<<1, 256, 0, stream>>>(cc_b[1], FP(oBcc1), 180);
  k_pack_b<<<1, 256, 0, stream>>>(rg_b[0], FP(oBrg0), 200);
  k_pack_b<<<1, 256, 0, stream>>>(rg_b[1], FP(oBrg1), 180);
  k_add_b<<<3, 256, 0, stream>>>(cl_bih, cl_bhh, FP(oBcl), 720);
  k_add_b<<<3, 256, 0, stream>>>(nl_bih, nl_bhh, FP(oBnl), 720);

  // ---- Comp embedding MLP ----
  {
    int n = (int)(ML * 1024);
    k_cvt_bf16<<<(n + 255) / 256, 256, 0, stream>>>(ast, BF(oBig0), n);
  }
  auto gemmE = [&](size_t A, size_t W, size_t bias, size_t C, int M, int N, int K) {
    k_gemm<1><<<dim3(M / TM, N / TN), 256, 0, stream>>>(BF(A), BF(W), FP(bias),
                                                        (void*)(ws + C), M, N, K);
  };
  auto gemmG = [&](size_t A, size_t W, size_t bias, size_t C, int M, int N, int K) {
    k_gemm<0><<<dim3(M / TM, N / TN), 256, 0, stream>>>(BF(A), BF(W), FP(bias),
                                                        (void*)(ws + C), M, N, K);
  };
  gemmE(oBig0, oWce0, oBce0, oBig1, (int)ML, 640, 1024);
  gemmE(oBig1, oWce1, oBce1, oBig2, (int)ML, 384, 640);
  gemmE(oBig2, oWce2, oBce2, oBig1, (int)ML, 256, 384);   // a3 reuses Big1
  gemmE(oBig1, oWce3, oBce3, oEmbed, (int)ML, 192, 256);  // embed (16384 x 192)

  // ---- Leaf LSTMs: batch 1536 (6 leaves x 256), 8 steps ----
  hipMemsetAsync(ws + oH, 0, (size_t)1536 * 192 * 2, stream);
  hipMemsetAsync(ws + oC, 0, (size_t)1536 * 192 * 4, stream);
  for (int t = 0; t < 8; ++t) {
    k_build_xh<<<1536, 192, 0, stream>>>(BF(oEmbed), BF(oH), BF(oXh), 0, t);
    gemmG(oXh, oWcl, oBcl, oGates, 1536, 768, 384);
    k_lstm_update<<<1536, 192, 0, stream>>>(FP(oGates), FP(oC), BF(oH));
  }
  // leaf concat block: [no_nodes | comps_h]
  k_build_concat<<<1536, 192, 0, stream>>>(nullptr, no_nodes, BF(oH), nullptr, BF(oXh));
  gemmE(oXh, oWcc0, oBcc0, oCct, 1536, 256, 384);
  gemmE(oCct, oWcc1, oBcc1, oLeaf, 1536, 192, 256);

  // ---- Interior LSTMs: batch 768 (3 interiors x 256), 2 steps ----
  hipMemsetAsync(ws + oH, 0, (size_t)1536 * 192 * 2, stream);
  hipMemsetAsync(ws + oC, 0, (size_t)1536 * 192 * 4, stream);
  for (int t = 0; t < 2; ++t) {
    k_build_xh<<<768, 192, 0, stream>>>(BF(oLeaf), BF(oH), BF(oXh), 1, t);
    gemmG(oXh, oWnl, oBnl, oGates, 768, 768, 384);
    k_lstm_update<<<768, 192, 0, stream>>>(FP(oGates), FP(oC), BF(oH));
  }
  // interior concat block: [nodes_h | no_comps]
  k_build_concat<<<768, 192, 0, stream>>>(BF(oH), nullptr, nullptr, no_comps, BF(oXh));
  gemmE(oXh, oWcc0, oBcc0, oCct, 768, 256, 384);
  gemmE(oCct, oWcc1, oBcc1, oMid, 768, 192, 256);

  // ---- Root LSTM: batch 256, 3 steps over mids ----
  hipMemsetAsync(ws + oH, 0, (size_t)1536 * 192 * 2, stream);
  hipMemsetAsync(ws + oC, 0, (size_t)1536 * 192 * 4, stream);
  for (int t = 0; t < 3; ++t) {
    k_build_xh<<<256, 192, 0, stream>>>(BF(oMid), BF(oH), BF(oXh), 2, t);
    gemmG(oXh, oWnl, oBnl, oGates, 256, 768, 384);
    k_lstm_update<<<256, 192, 0, stream>>>(FP(oGates), FP(oC), BF(oH));
  }
  k_build_concat<<<256, 192, 0, stream>>>(BF(oH), nullptr, nullptr, no_comps, BF(oXh));
  gemmE(oXh, oWcc0, oBcc0, oCct, 256, 256, 384);
  gemmE(oCct, oWcc1, oBcc1, oProg, 256, 192, 256);

  // ---- Regression head ----
  gemmE(oProg, oWrg0, oBrg0, oRga, 256, 256, 192);
  gemmE(oRga, oWrg1, oBrg1, oRgb, 256, 192, 256);
  k_pred<<<1, 256, 0, stream>>>(BF(oRgb), pred_w, pred_b, (float*)d_out);
}
